// ExternalEmbeddingSelfAttention_55121610277508
// MI455X (gfx1250) — matrix operand order repaired and checked by reading, not compiled, not run on hardware
//
#include <hip/hip_runtime.h>

// ExternalEmbeddingSelfAttention for MI455X (gfx1250, wave32, WMMA).
// bf16 WMMA (v_wmma_f32_16x16x32_bf16) with fp32 accumulation; fused
// Q/K/V projection + scores + softmax + context mixing.
// M=32 rows per wave: each weight B-fragment feeds 2 WMMAs (halves L2 traffic).
// A-tile staged via async global->LDS (ASYNCcnt) when available.

#define Hh 768
#define Bz 4
#define Ss 4096
#define Ee 32

typedef __attribute__((ext_vector_type(16))) __bf16 v16bf;
typedef __attribute__((ext_vector_type(8)))  float  v8f;
typedef __attribute__((ext_vector_type(4)))  int    v4i;

union FragU { uint4 u[2]; v16bf v; };

__device__ __forceinline__ v16bf load_frag2(const __bf16* p0, const __bf16* p1) {
  FragU f;
  f.u[0] = *(const uint4*)p0;
  f.u[1] = *(const uint4*)p1;
  return f.v;
}

__device__ __forceinline__ v8f wmma_bf16(v16bf a, v16bf b, v8f c) {
  // (neg_a, A, neg_b, B, c_mod, C, reuse_a, reuse_b)
  return __builtin_amdgcn_wmma_f32_16x16x32_bf16(false, a, false, b, (short)0, c,
                                                 false, false);
}

// Stage n uint4's from global to LDS; async-to-LDS path if toolchain has it.
__device__ __forceinline__ void stage_tile(const uint4* __restrict__ g,
                                           uint4* __restrict__ s,
                                           int n, int lane) {
#if __has_builtin(__builtin_amdgcn_global_load_async_to_lds_b128)
  typedef __attribute__((address_space(1))) v4i gv4i;
  typedef __attribute__((address_space(3))) v4i lv4i;
  for (int i = lane; i < n; i += 32) {
    __builtin_amdgcn_global_load_async_to_lds_b128(
        (gv4i*)(unsigned long long)(uintptr_t)(g + i),
        (lv4i*)(unsigned int)(uintptr_t)(s + i),
        0, 0);
  }
#if __has_builtin(__builtin_amdgcn_s_wait_asynccnt)
  __builtin_amdgcn_s_wait_asynccnt(0);
#else
  asm volatile("s_wait_asynccnt 0x0" ::: "memory");
#endif
#else
  for (int i = lane; i < n; i += 32) s[i] = g[i];
#endif
}

// ---------------- prep kernels ----------------

__global__ void cvt_bf16_kernel(const float* __restrict__ in, __bf16* __restrict__ out, int n) {
  int i = blockIdx.x * blockDim.x + threadIdx.x;
  if (i < n) out[i] = (__bf16)in[i];
}

// Wt[n][k] = (bf16) W[k][n]  (B-fragments read contiguous K per output column)
__global__ void transpose_w_kernel(const float* __restrict__ W, __bf16* __restrict__ Wt) {
  int i = blockIdx.x * blockDim.x + threadIdx.x;
  if (i >= Hh * Hh) return;
  int n = i / Hh;
  int k = i - n * Hh;
  Wt[i] = (__bf16)W[k * Hh + n];
}

// Kx[b][e][k] (contiguous k) and Vxt[b][n][e] (contiguous e), biased, bf16.
__global__ void kv_ext_kernel(const float* __restrict__ ext,
                              const float* __restrict__ Wk, const float* __restrict__ bk,
                              const float* __restrict__ Wv, const float* __restrict__ bv,
                              __bf16* __restrict__ Kxb, __bf16* __restrict__ Vxt) {
  int i = blockIdx.x * blockDim.x + threadIdx.x;   // over B*E*H
  if (i >= Bz * Ee * Hh) return;
  int n  = i % Hh;
  int be = i / Hh;                // b*E + e
  int b  = be / Ee;
  int e  = be - b * Ee;
  const float* x = ext + (size_t)be * Hh;
  float kx = bk[n], vx = bv[n];
  for (int h = 0; h < Hh; ++h) {
    float xv = x[h];
    kx += xv * Wk[h * Hh + n];
    vx += xv * Wv[h * Hh + n];
  }
  Kxb[(size_t)be * Hh + n] = (__bf16)kx;
  Vxt[((size_t)b * Hh + n) * Ee + e] = (__bf16)vx;
}

// ---------------- fused main kernel ----------------
// 1 wave / block, wave owns 32 rows (2 row-tiles). LDS layout (57728 B):
//   At     bf16 [32][768]   @ 0       (49152 B)
//   QA     bf16 [32][32]    @ 49152   ( 2048 B)
//   scores f32  [32][33]    @ 51200   ( 4224 B)
//   sSelf  f32  [32]        @ 55424   (  128 B)
//   p0s    f32  [32]        @ 55552   (  128 B)
//   Pw     bf16 [32][32]    @ 55680   ( 2048 B)
#define SMEM_BYTES 57728

__global__ void __launch_bounds__(32)
fused_attn_kernel(const __bf16* __restrict__ hidB,
                  const __bf16* __restrict__ Wqt,
                  const __bf16* __restrict__ Wkt,
                  const __bf16* __restrict__ Wvt,
                  const __bf16* __restrict__ Kxb,
                  const __bf16* __restrict__ Vxt,
                  const float* __restrict__ bq,
                  const float* __restrict__ bk,
                  const float* __restrict__ bv,
                  const float* __restrict__ doclog,
                  float* __restrict__ out) {
  extern __shared__ char smem[];
  const int lane = threadIdx.x & 31;
  const int half = lane >> 4;
  const int l16  = lane & 15;

  const int tileIdx = blockIdx.x;                  // 0..511
  const int b    = tileIdx >> 7;                    // / (S/32)
  const int row0 = (tileIdx & 127) << 5;

  __bf16* At     = (__bf16*)smem;
  __bf16* QA     = (__bf16*)(smem + 49152);
  float*  scores = (float*)(smem + 51200);
  float*  sSelf  = (float*)(smem + 55424);
  float*  p0s    = (float*)(smem + 55552);
  __bf16* Pw     = (__bf16*)(smem + 55680);

  // Phase 0: stage 32x768 bf16 hidden tile into LDS (async path).
  stage_tile((const uint4*)(hidB + (size_t)(b * Ss + row0) * Hh),
             (uint4*)At, 32 * Hh / 8, lane);
  __syncthreads();

  v8f sx[2][2] = {};
  float sse[2][8] = {};
  const __bf16* arow0 = At + l16 * Hh;
  const __bf16* arow1 = At + (16 + l16) * Hh;

  // Phase 1: Q/K chunks (32 cols), self score, external scores.
#pragma unroll 1
  for (int c0 = 0; c0 < Hh; c0 += 32) {
    v8f q[2][2] = {};
    v8f ka[2][2] = {};
#pragma unroll 1
    for (int kk = 0; kk < Hh; kk += 32) {
      v16bf a0 = load_frag2(arow0 + kk + half * 16, arow0 + kk + half * 16 + 8);
      v16bf a1 = load_frag2(arow1 + kk + half * 16, arow1 + kk + half * 16 + 8);
      const __bf16* wq0 = Wqt + (size_t)(c0 + l16) * Hh + kk + half * 16;
      const __bf16* wq1 = wq0 + 16 * Hh;
      const __bf16* wk0 = Wkt + (size_t)(c0 + l16) * Hh + kk + half * 16;
      const __bf16* wk1 = wk0 + 16 * Hh;
      v16bf fq0 = load_frag2(wq0, wq0 + 8);
      v16bf fq1 = load_frag2(wq1, wq1 + 8);
      v16bf fk0 = load_frag2(wk0, wk0 + 8);
      v16bf fk1 = load_frag2(wk1, wk1 + 8);
      q[0][0]  = wmma_bf16(a0, fq0, q[0][0]);
      q[1][0]  = wmma_bf16(a1, fq0, q[1][0]);
      q[0][1]  = wmma_bf16(a0, fq1, q[0][1]);
      q[1][1]  = wmma_bf16(a1, fq1, q[1][1]);
      ka[0][0] = wmma_bf16(a0, fk0, ka[0][0]);
      ka[1][0] = wmma_bf16(a1, fk0, ka[1][0]);
      ka[0][1] = wmma_bf16(a0, fk1, ka[0][1]);
      ka[1][1] = wmma_bf16(a1, fk1, ka[1][1]);
    }
    float bqn0 = bq[c0 + l16], bqn1 = bq[c0 + 16 + l16];
    float bkn0 = bk[c0 + l16], bkn1 = bk[c0 + 16 + l16];
#pragma unroll
    for (int r = 0; r < 2; ++r) {
#pragma unroll
      for (int v = 0; v < 8; ++v) {
        q[r][0][v] += bqn0; q[r][1][v] += bqn1;
        ka[r][0][v] += bkn0; ka[r][1][v] += bkn1;
        sse[r][v] += q[r][0][v] * ka[r][0][v] + q[r][1][v] * ka[r][1][v];
      }
    }
    // Re-pack Q chunk as bf16 A-fragments via LDS.
#pragma unroll
    for (int r = 0; r < 2; ++r) {
#pragma unroll
      for (int v = 0; v < 8; ++v) {
        int m = r * 16 + v + 8 * half;
        QA[m * 32 + l16]      = (__bf16)q[r][0][v];
        QA[m * 32 + 16 + l16] = (__bf16)q[r][1][v];
      }
    }
    __syncthreads();
    {
      const __bf16* qr0 = QA + l16 * 32;
      const __bf16* qr1 = QA + (16 + l16) * 32;
      v16bf aq0 = load_frag2(qr0 + half * 16, qr0 + half * 16 + 8);
      v16bf aq1 = load_frag2(qr1 + half * 16, qr1 + half * 16 + 8);
      const __bf16* kx0 = Kxb + (size_t)(b * Ee + l16) * Hh + c0 + half * 16;
      const __bf16* kx1 = Kxb + (size_t)(b * Ee + 16 + l16) * Hh + c0 + half * 16;
      v16bf fx0 = load_frag2(kx0, kx0 + 8);
      v16bf fx1 = load_frag2(kx1, kx1 + 8);
      sx[0][0] = wmma_bf16(aq0, fx0, sx[0][0]);
      sx[0][1] = wmma_bf16(aq0, fx1, sx[0][1]);
      sx[1][0] = wmma_bf16(aq1, fx0, sx[1][0]);
      sx[1][1] = wmma_bf16(aq1, fx1, sx[1][1]);
    }
    __syncthreads();
  }

  // Reduce s_self over the 16 lanes of each half (n dimension).
#pragma unroll
  for (int r = 0; r < 2; ++r) {
#pragma unroll
    for (int v = 0; v < 8; ++v) {
      float x = sse[r][v];
      for (int off = 1; off < 16; off <<= 1) x += __shfl_xor(x, off, 32);
      sse[r][v] = x;
    }
  }
  if (l16 == 0) {
#pragma unroll
    for (int r = 0; r < 2; ++r)
#pragma unroll
      for (int v = 0; v < 8; ++v) sSelf[r * 16 + v + 8 * half] = sse[r][v];
  }
#pragma unroll
  for (int r = 0; r < 2; ++r) {
#pragma unroll
    for (int v = 0; v < 8; ++v) {
      int m = r * 16 + v + 8 * half;
      scores[m * 33 + 1 + l16]  = sx[r][0][v];
      scores[m * 33 + 17 + l16] = sx[r][1][v];
    }
  }
  __syncthreads();

  // Softmax over 33 scores (one row per lane), fold doc_logprobs (bf16).
  {
    int m = lane;
    float* sr = scores + m * 33;
    sr[0] = sSelf[m];
    float mx = -1e30f;
    for (int i = 0; i < 33; ++i) mx = fmaxf(mx, sr[i]);
    float sum = 0.f;
    for (int i = 0; i < 33; ++i) sum += __expf(sr[i] - mx);
    float inv = 1.f / sum;
    p0s[m] = __expf(sr[0] - mx) * inv;
    for (int e = 0; e < Ee; ++e)
      Pw[m * 32 + e] = (__bf16)(__expf(sr[1 + e] - mx) * inv * doclog[b * Ee + e]);
  }
  __syncthreads();

  // Phase 2: V tiles + context mixing.
  float p0r[2][8];
#pragma unroll
  for (int r = 0; r < 2; ++r)
#pragma unroll
    for (int v = 0; v < 8; ++v) p0r[r][v] = p0s[r * 16 + v + 8 * half];
  const __bf16* pr0 = Pw + l16 * 32;
  const __bf16* pr1 = Pw + (16 + l16) * 32;
  v16bf apw0 = load_frag2(pr0 + half * 16, pr0 + half * 16 + 8);
  v16bf apw1 = load_frag2(pr1 + half * 16, pr1 + half * 16 + 8);

#pragma unroll 1
  for (int n0 = 0; n0 < Hh; n0 += 16) {
    v8f vacc0 = {}, vacc1 = {};
    const __bf16* wv = Wvt + (size_t)(n0 + l16) * Hh;
#pragma unroll 2
    for (int kk = 0; kk < Hh; kk += 32) {
      v16bf a0 = load_frag2(arow0 + kk + half * 16, arow0 + kk + half * 16 + 8);
      v16bf a1 = load_frag2(arow1 + kk + half * 16, arow1 + kk + half * 16 + 8);
      const __bf16* wvp = wv + kk + half * 16;
      v16bf fv = load_frag2(wvp, wvp + 8);
      vacc0 = wmma_bf16(a0, fv, vacc0);
      vacc1 = wmma_bf16(a1, fv, vacc1);
    }
    float bvn = bv[n0 + l16];
    v8f o0, o1;
#pragma unroll
    for (int v = 0; v < 8; ++v) {
      o0[v] = p0r[0][v] * (vacc0[v] + bvn);
      o1[v] = p0r[1][v] * (vacc1[v] + bvn);
    }
    const __bf16* vx = Vxt + (size_t)(b * Hh + n0 + l16) * Ee + half * 16;
    v16bf bvx = load_frag2(vx, vx + 8);
    o0 = wmma_bf16(apw0, bvx, o0);   // K=32 over E
    o1 = wmma_bf16(apw1, bvx, o1);
    float* op = out + (size_t)(b * Ss + row0) * Hh + n0 + l16;
#pragma unroll
    for (int v = 0; v < 8; ++v) {
      op[(size_t)(v + 8 * half) * Hh]      = o0[v];
      op[(size_t)(16 + v + 8 * half) * Hh] = o1[v];
    }
  }
}

// ---------------- launch ----------------

extern "C" void kernel_launch(void* const* d_in, const int* in_sizes, int n_in,
                              void* d_out, int out_size, void* d_ws, size_t ws_size,
                              hipStream_t stream) {
  const float* hidden = (const float*)d_in[0];
  const float* ext    = (const float*)d_in[1];
  const float* doclog = (const float*)d_in[2];
  const float* Wq     = (const float*)d_in[3];
  const float* bq     = (const float*)d_in[4];
  const float* Wk     = (const float*)d_in[5];
  const float* bk     = (const float*)d_in[6];
  const float* Wv     = (const float*)d_in[7];
  const float* bv     = (const float*)d_in[8];
  float* out = (float*)d_out;

  char* ws = (char*)d_ws;
  __bf16* hidB = (__bf16*)ws; ws += (size_t)Bz * Ss * Hh * 2;
  __bf16* Wqt  = (__bf16*)ws; ws += (size_t)Hh * Hh * 2;
  __bf16* Wkt  = (__bf16*)ws; ws += (size_t)Hh * Hh * 2;
  __bf16* Wvt  = (__bf16*)ws; ws += (size_t)Hh * Hh * 2;
  __bf16* Kxb  = (__bf16*)ws; ws += (size_t)Bz * Ee * Hh * 2;
  __bf16* Vxt  = (__bf16*)ws; ws += (size_t)Bz * Hh * Ee * 2;

  int nh = Bz * Ss * Hh;
  cvt_bf16_kernel<<<(nh + 255) / 256, 256, 0, stream>>>(hidden, hidB, nh);
  transpose_w_kernel<<<(Hh * Hh + 255) / 256, 256, 0, stream>>>(Wq, Wqt);
  transpose_w_kernel<<<(Hh * Hh + 255) / 256, 256, 0, stream>>>(Wk, Wkt);
  transpose_w_kernel<<<(Hh * Hh + 255) / 256, 256, 0, stream>>>(Wv, Wvt);
  kv_ext_kernel<<<(Bz * Ee * Hh + 255) / 256, 256, 0, stream>>>(ext, Wk, bk, Wv, bv, Kxb, Vxt);

  fused_attn_kernel<<<Bz * Ss / 32, 32, SMEM_BYTES, stream>>>(
      hidB, Wqt, Wkt, Wvt, Kxb, Vxt, bq, bk, bv, doclog, out);
}
